// MoELoRALinear_7954279432696
// MI455X (gfx1250) — matrix operand order repaired and checked by reading, not compiled, not run on hardware
//
#include <hip/hip_runtime.h>
#include <hip/hip_bf16.h>
#include <stdint.h>

// ---------------------------------------------------------------------------
// MoE-LoRA fused linear for gfx1250 (wave32)
//  - pass 1: fp32 -> bf16 conversion of x / W_base / B^T into workspace
//  - pass 2: router softmax + top-2 gate + low-rank h = x.A -> bf16 coeffs
//  - pass 3: WMMA bf16 GEMM (f32 accum); tiles staged by the Tensor Data
//            Mover (tensor_load_to_lds, TENSORcnt) with a 3-deep pipeline;
//            LoRA rank-32 tail fused as a 33rd K-step
// ---------------------------------------------------------------------------

typedef __attribute__((ext_vector_type(16))) __bf16 v16bf;
typedef __attribute__((ext_vector_type(8)))  float  v8f;
typedef __attribute__((ext_vector_type(4)))  unsigned int v4u;
typedef __attribute__((ext_vector_type(8)))  int v8i;
typedef __attribute__((ext_vector_type(4)))  int v4i;

#define D_DIM 1024
#define E_NUM 8
#define ER    32          // E*R = rank of fused LoRA tail GEMM
#define NTOK  16384       // Bt*S (fixed by harness)

// GEMM tiling
#define BM 128
#define BN 128
#define BK 32
#define XS_STRIDE 40      // ushort units per LDS row (64B data + 16B TDM pad)
#define NKT (D_DIM / BK + 1)   // 32 main K-steps + 1 LoRA tail step
#define NBUF 3                 // TDM pipeline depth

union FragU {
    uint4 u4[2];
    v16bf v;
};

__device__ __forceinline__ unsigned short f2bf(float f) {
    union { float f; unsigned int u; } c;
    c.f = f;
    unsigned int u = c.u;
    // round-to-nearest-even (inputs are finite gaussians; skip NaN path)
    return (unsigned short)((u + 0x7FFFu + ((u >> 16) & 1u)) >> 16);
}

__device__ __forceinline__ uint2 pack4bf(float4 v) {
    union { unsigned short us[4]; uint2 u2; } p;
    p.us[0] = f2bf(v.x); p.us[1] = f2bf(v.y);
    p.us[2] = f2bf(v.z); p.us[3] = f2bf(v.w);
    return p.u2;
}

// ---------------------------------------------------------------------------
// TDM: issue a 2D bf16 tile load (tile_d0 x tile_d1 elements, row stride
// d0_stride elements) into LDS with 16B padding appended per 64B row
// (pad_interval = 16 DWORDs, pad_amount = 4 DWORDs -> 80B LDS rows).
// Descriptor layout per CDNA5 ISA 8.3/8.4. 6-arg builtin (clang-23 arity).
// ---------------------------------------------------------------------------
__device__ __forceinline__ void tdm_load_2d(unsigned lds_off, const void* gaddr,
                                            unsigned tensor_d0, unsigned tensor_d1,
                                            unsigned d0_stride,
                                            unsigned tile_d0, unsigned tile_d1) {
    const unsigned long long ga = (unsigned long long)(uintptr_t)gaddr;
    v4u g0;
    g0[0] = 1u;                                            // count=1 (valid, user)
    g0[1] = lds_off;                                       // lds_addr [63:32]
    g0[2] = (unsigned)(ga & 0xffffffffu);                  // global_addr lo
    g0[3] = (unsigned)((ga >> 32) & 0x01ffffffu)           // global_addr hi (57b)
          | (2u << 30);                                    // type=2 ("image")
    v8i g1;
    g1[0] = (int)((1u << 16)                               // data_size = 2 bytes
                | (1u << 20)                               // pad_enable
                | (3u << 22)                               // pad_interval: 16 DW
                | (3u << 25));                             // pad_amount: 4 DW
    g1[1] = (int)((tensor_d0 & 0xffffu) << 16);            // td0[15:0] @ [63:48]
    g1[2] = (int)((tensor_d0 >> 16) | ((tensor_d1 & 0xffffu) << 16));
    g1[3] = (int)((tensor_d1 >> 16) | (tile_d0 << 16));    // tile_dim0 @ [127:112]
    g1[4] = (int)(tile_d1 & 0xffffu);                      // tile_dim1; tile_dim2=0
    g1[5] = (int)d0_stride;                                // tensor_dim0_stride lo
    g1[6] = 0;                                             // stride hi, dim1 stride
    g1[7] = 0;
    const v4i z4 = {0, 0, 0, 0};                           // 2D: groups 2/3 unused
    const v8i z8 = {0, 0, 0, 0, 0, 0, 0, 0};
    __builtin_amdgcn_tensor_load_to_lds(g0, g1, z4, z4, z8, 0);
}

// ---------------------------------------------------------------------------
// Kernel 1: one-time fp32 -> bf16 conversion (x, W_base row-major; B transposed
// to [out][er] so the LoRA tail tile has the same layout as W tiles).
// ---------------------------------------------------------------------------
__global__ void cvt_bf16_kernel(const float* __restrict__ x,
                                const float* __restrict__ Wb,
                                const float* __restrict__ Bm,
                                unsigned short* __restrict__ xbf,
                                unsigned short* __restrict__ wbf,
                                unsigned short* __restrict__ Bt) {
    const int tid = blockIdx.x * blockDim.x + threadIdx.x;
    const int nth = gridDim.x * blockDim.x;

    const float4* x4 = (const float4*)x;
    uint2* xo = (uint2*)xbf;
    const int nx4 = NTOK * D_DIM / 4;
    for (int i = tid; i < nx4; i += nth) xo[i] = pack4bf(x4[i]);

    const float4* w4 = (const float4*)Wb;
    uint2* wo = (uint2*)wbf;
    const int nw4 = D_DIM * D_DIM / 4;
    for (int i = tid; i < nw4; i += nth) wo[i] = pack4bf(w4[i]);

    // Bt[n][k] = B_flat[k][n],  k = e*4+r in [0,32), n in [0,1024)
    for (int i = tid; i < D_DIM * ER; i += nth) {
        const int n = i >> 5, k = i & 31;
        Bt[i] = f2bf(Bm[(size_t)k * D_DIM + n]);
    }
}

// ---------------------------------------------------------------------------
// Kernel 2: router softmax + top-2 gate + low-rank h = x.A, gated in low-rank
// space. One wave per token; writes 32 bf16 coefficients per token.
// ---------------------------------------------------------------------------
__global__ void moe_router_coeff(const float* __restrict__ x,
                                 const float* __restrict__ Wr,
                                 const float* __restrict__ A,
                                 unsigned short* __restrict__ coeff,
                                 int n_tokens) {
    const int lane  = threadIdx.x & 31;
    const int wave  = threadIdx.x >> 5;
    const int token = blockIdx.x * 8 + wave;
    if (token >= n_tokens) return;

    const float*  xr = x + (size_t)token * D_DIM;
    const float4* A4 = (const float4*)A;   // [E*D] float4 over r-dimension

    float lg[E_NUM];
    float h[ER];
#pragma unroll
    for (int e = 0; e < E_NUM; ++e) lg[e] = 0.f;
#pragma unroll
    for (int i = 0; i < ER; ++i) h[i] = 0.f;

    for (int d = lane; d < D_DIM; d += 32) {
        float xv = xr[d];
#pragma unroll
        for (int e = 0; e < E_NUM; ++e) {
            lg[e] = fmaf(xv, Wr[e * D_DIM + d], lg[e]);
            float4 a4 = A4[e * D_DIM + d];
            h[e * 4 + 0] = fmaf(xv, a4.x, h[e * 4 + 0]);
            h[e * 4 + 1] = fmaf(xv, a4.y, h[e * 4 + 1]);
            h[e * 4 + 2] = fmaf(xv, a4.z, h[e * 4 + 2]);
            h[e * 4 + 3] = fmaf(xv, a4.w, h[e * 4 + 3]);
        }
    }

    // Butterfly reduction: all lanes end up with the full sums.
#pragma unroll
    for (int off = 16; off > 0; off >>= 1) {
#pragma unroll
        for (int e = 0; e < E_NUM; ++e) lg[e] += __shfl_xor(lg[e], off, 32);
#pragma unroll
        for (int i = 0; i < ER; ++i) h[i] += __shfl_xor(h[i], off, 32);
    }

    // Softmax over experts (uniform across lanes).
    float m = lg[0];
#pragma unroll
    for (int e = 1; e < E_NUM; ++e) m = fmaxf(m, lg[e]);
    float g[E_NUM];
    float s = 0.f;
#pragma unroll
    for (int e = 0; e < E_NUM; ++e) { g[e] = __expf(lg[e] - m); s += g[e]; }
    float inv = 1.0f / s;
#pragma unroll
    for (int e = 0; e < E_NUM; ++e) g[e] *= inv;

    // Top-2 (ties prefer lower index, matching lax.top_k).
    int i1 = 0; float v1 = g[0];
#pragma unroll
    for (int e = 1; e < E_NUM; ++e) { if (g[e] > v1) { v1 = g[e]; i1 = e; } }
    int i2 = -1; float v2 = -1.f;
#pragma unroll
    for (int e = 0; e < E_NUM; ++e) { if (e != i1 && g[e] > v2) { v2 = g[e]; i2 = e; } }

    float denom = 1.0f / (v1 + v2 + 1e-6f);
    float gs[E_NUM];
#pragma unroll
    for (int e = 0; e < E_NUM; ++e)
        gs[e] = (e == i1 || e == i2) ? g[e] * denom : 0.f;

    // Lane l writes coefficient l (uniform unrolled cndmask chain).
    float val = 0.f;
#pragma unroll
    for (int j = 0; j < ER; ++j) {
        float cj = h[j] * gs[j >> 2];
        val = (lane == j) ? cj : val;
    }
    coeff[(size_t)token * ER + lane] = f2bf(val);
}

// ---------------------------------------------------------------------------
// Kernel 3: out = x @ W_base^T + b_base + coeff @ B_flat   (all bf16 operands)
// 128x128 tile / 256 threads (8 waves); wave tile 64x32 => 8 wmma accums.
// Wave 0 drives the TDM, 3 tiles in flight; TENSORcnt retires in order so
// waiting <=2 guarantees the tile being consumed is resident.
// ---------------------------------------------------------------------------
__global__ void __launch_bounds__(256)
moe_base_lora_gemm(const unsigned short* __restrict__ xbf,
                   const unsigned short* __restrict__ wbf,
                   const unsigned short* __restrict__ coeff,
                   const unsigned short* __restrict__ Bt,
                   const float* __restrict__ bb,
                   float* __restrict__ out) {
    __shared__ __align__(16) unsigned short xs[NBUF][BM * XS_STRIDE];
    __shared__ __align__(16) unsigned short wsm[NBUF][BN * XS_STRIDE];

    const int t    = threadIdx.x;
    const int lane = t & 31;
    const int wave = t >> 5;
    const int wrow = wave >> 2;        // 0..1 -> 64-row slab
    const int wcol = wave & 3;         // 0..3 -> 32-col slab
    const int half = lane >> 4;        // K-half selector
    const int l15  = lane & 15;

    const int m0 = blockIdx.y * BM;
    const int n0 = blockIdx.x * BN;

    const unsigned short* xtile = xbf   + (size_t)m0 * D_DIM;  // [128][1024]
    const unsigned short* wtile = wbf   + (size_t)n0 * D_DIM;
    const unsigned short* ctile = coeff + (size_t)m0 * ER;     // [128][32]
    const unsigned short* btile = Bt    + (size_t)n0 * ER;

    unsigned lx[NBUF], lw[NBUF];
#pragma unroll
    for (int b = 0; b < NBUF; ++b) {
        lx[b] = (unsigned)(uintptr_t)(&xs[b][0]);
        lw[b] = (unsigned)(uintptr_t)(&wsm[b][0]);
    }

    v8f acc[4][2] = {};

    // Prologue: wave 0 stages tiles 0 and 1 via the Tensor Data Mover.
    if (wave == 0) {
        tdm_load_2d(lx[0], xtile, D_DIM, BM, D_DIM, BK, BM);
        tdm_load_2d(lw[0], wtile, D_DIM, BN, D_DIM, BK, BN);
        tdm_load_2d(lx[1], xtile + BK, D_DIM, BM, D_DIM, BK, BM);
        tdm_load_2d(lw[1], wtile + BK, D_DIM, BN, D_DIM, BK, BN);
    }

    for (int ks = 0; ks < NKT; ++ks) {
        if (wave == 0) {
            // tile ks resident when <= 2 TDM ops (tile ks+1) remain outstanding
            if (ks < NKT - 1) __builtin_amdgcn_s_wait_tensorcnt(2);
            else              __builtin_amdgcn_s_wait_tensorcnt(0);
        }
        __syncthreads();   // publish tile ks; buffer (ks+2)%3 free to overwrite

        if (wave == 0) {
            const int nxt2 = ks + 2;
            const int b = (nxt2 >= NBUF) ? (nxt2 - NBUF) : nxt2;  // = nxt2 % 3 only for nxt2 < 6
            if (nxt2 < NKT - 1) {
                tdm_load_2d(lx[b], xtile + nxt2 * BK, D_DIM, BM, D_DIM, BK, BM);
                tdm_load_2d(lw[b], wtile + nxt2 * BK, D_DIM, BN, D_DIM, BK, BN);
            } else if (nxt2 == NKT - 1) {
                // LoRA tail: same tile shape, 64B rows (stride = 32 elements)
                tdm_load_2d(lx[b], ctile, ER, BM, ER, BK, BM);
                tdm_load_2d(lw[b], btile, ER, BN, ER, BK, BN);
            }
        }

        const int cb = ks % NBUF;
        const unsigned short* xsb = xs[cb];
        const unsigned short* wsb = wsm[cb];

        // Build fragments from LDS (layouts per CDNA5 ISA 7.12.2).
        FragU afrag[4];
#pragma unroll
        for (int tm = 0; tm < 4; ++tm) {
            const int row = wrow * 64 + tm * 16 + l15;
            const unsigned short* p = &xsb[row * XS_STRIDE + half * 8];
            afrag[tm].u4[0] = *(const uint4*)(p);       // K {0..7} / {8..15}
            afrag[tm].u4[1] = *(const uint4*)(p + 16);  // K {16..23} / {24..31}
        }
        FragU bfrag[2];
#pragma unroll
        for (int tn = 0; tn < 2; ++tn) {
            const int nrow = wcol * 32 + tn * 16 + l15;
            const unsigned short* p = &wsb[nrow * XS_STRIDE + half * 8];
            bfrag[tn].u4[0] = *(const uint4*)(p);       // K {0..7} / {8..15}
            bfrag[tn].u4[1] = *(const uint4*)(p + 16);  // K {16..23} / {24..31}
        }

#pragma unroll
        for (int tm = 0; tm < 4; ++tm)
#pragma unroll
            for (int tn = 0; tn < 2; ++tn)
                acc[tm][tn] = __builtin_amdgcn_wmma_f32_16x16x32_bf16(
                    false, afrag[tm].v, false, bfrag[tn].v,
                    (short)0, acc[tm][tn], false, false);
    }

    // Epilogue: + bias, store fp32.
#pragma unroll
    for (int tn = 0; tn < 2; ++tn) {
        const int col  = n0 + wcol * 32 + tn * 16 + l15;
        const float bv = bb[col];
#pragma unroll
        for (int tm = 0; tm < 4; ++tm) {
            const int rowb = m0 + wrow * 64 + tm * 16 + half * 8;
#pragma unroll
            for (int v = 0; v < 8; ++v) {
                out[(size_t)(rowb + v) * D_DIM + col] = acc[tm][tn][v] + bv;
            }
        }
    }
}

// ---------------------------------------------------------------------------
extern "C" void kernel_launch(void* const* d_in, const int* in_sizes, int n_in,
                              void* d_out, int out_size, void* d_ws, size_t ws_size,
                              hipStream_t stream) {
    const float* x  = (const float*)d_in[0];   // [Bt,S,D]
    const float* Wb = (const float*)d_in[1];   // [D,D] (out, in)
    const float* bb = (const float*)d_in[2];   // [D]
    const float* Wr = (const float*)d_in[3];   // [E,D]
    const float* A  = (const float*)d_in[4];   // [E,D,R]
    const float* Bm = (const float*)d_in[5];   // [E,R,D]
    float* out = (float*)d_out;

    const int n_tokens = in_sizes[0] / D_DIM;  // 16384

    // Workspace layout (bf16 staging), ~35 MB total.
    char* ws = (char*)d_ws;
    unsigned short* xbf   = (unsigned short*)(ws);                               // 32 MB
    unsigned short* wbf   = (unsigned short*)(ws + (size_t)NTOK * D_DIM * 2);    // 2 MB
    unsigned short* Bt    = (unsigned short*)(ws + (size_t)(NTOK + D_DIM) * D_DIM * 2); // 64 KB
    unsigned short* coeff = Bt + (size_t)D_DIM * ER;                             // 1 MB

    cvt_bf16_kernel<<<dim3(2048), dim3(256), 0, stream>>>(x, Wb, Bm, xbf, wbf, Bt);

    moe_router_coeff<<<dim3((n_tokens + 7) / 8), dim3(256), 0, stream>>>(
        x, Wr, A, coeff, n_tokens);

    dim3 grid(D_DIM / BN, n_tokens / BM);
    moe_base_lora_gemm<<<grid, dim3(256), 0, stream>>>(
        xbf, wbf, coeff, Bt, bb, out);
}
